// SparseMoE_33724083208853
// MI455X (gfx1250) — matrix operand order repaired and checked by reading, not compiled, not run on hardware
//
#include <hip/hip_runtime.h>
#include <hip/hip_bf16.h>
#include <math.h>

typedef _Float16 f16;
typedef __attribute__((ext_vector_type(16))) _Float16 v16h;
typedef __attribute__((ext_vector_type(8)))  _Float16 v8h;
typedef __attribute__((ext_vector_type(8)))  float    v8f;

#define Dm 1024
#define Hm 512
#define Em 8
#define NTOK 16384
#define NPAIR 28
#define CAPm 16384
#define MAXTILES 1024
#define KBD 32   /* D/32 k-blocks */
#define KBH 16   /* H/32 k-blocks */

static constexpr size_t OFF_CNT = 0;
static constexpr size_t OFF_TOK = 256;
static constexpr size_t SZ_LIST = (size_t)NPAIR * CAPm * 4;      // 1,835,008 B
static constexpr size_t OFF_GLO = OFF_TOK + SZ_LIST;
static constexpr size_t OFF_GHI = OFF_GLO + SZ_LIST;
static constexpr size_t OFF_WDP = OFF_GHI + SZ_LIST;
static constexpr size_t NWD = (size_t)Em * KBD * Hm * 32;        // 4,194,304 f16
static constexpr size_t NWU = (size_t)Em * KBH * Dm * 32;        // 4,194,304 f16
static constexpr size_t OFF_WUP = OFF_WDP + NWD * 2;

union AFrag { v16h v; v8h h[2]; };

__global__ void init_k(int* cnt) {
  if (threadIdx.x < NPAIR) cnt[threadIdx.x] = 0;
}

// Pack Wd [E][D][H] -> WdP[e][kb][h][kl], Wu [E][H][D] -> WuP[e][kb][d][kl] (f16, kl = k%32)
__global__ void __launch_bounds__(256) pack_k(const float* __restrict__ Wd,
                                              const float* __restrict__ Wu,
                                              f16* __restrict__ WdP, f16* __restrict__ WuP) {
  unsigned i = blockIdx.x * 256u + threadIdx.x;
  if (i < (unsigned)NWD) {
    unsigned kl = i & 31u, nH = (i >> 5) & (Hm - 1), kb = (i >> 14) & 31u, e = i >> 19;
    WdP[i] = (f16)Wd[((size_t)e * Dm + (kb * 32 + kl)) * Hm + nH];
  } else {
    unsigned j = i - (unsigned)NWD;
    unsigned kl = j & 31u, nD = (j >> 5) & (Dm - 1), kb = (j >> 15) & 15u, e = j >> 19;
    WuP[j] = (f16)Wu[((size_t)e * Hm + (kb * 32 + kl)) * Dm + nD];
  }
}

// One wave per token: routing logits, noisy top-2, gates, bucket by expert pair.
__global__ void __launch_bounds__(256) router_k(
    const float* __restrict__ x, const float* __restrict__ Wr, const float* __restrict__ br,
    const float* __restrict__ Wn, const float* __restrict__ bn, const float* __restrict__ noise,
    int* __restrict__ cnt, int* __restrict__ tokL,
    float* __restrict__ gloL, float* __restrict__ ghiL) {
  const int lane = threadIdx.x & 31;
  const int t = blockIdx.x * 8 + (threadIdx.x >> 5);
  float aR[Em], aN[Em];
  #pragma unroll
  for (int e = 0; e < Em; ++e) { aR[e] = 0.f; aN[e] = 0.f; }
  const float* xr = x + (size_t)t * Dm;
  for (int i = 0; i < Dm / 32; ++i) {
    int d = lane + i * 32;
    float xv = xr[d];
    const float* wr = Wr + d * Em;
    const float* wn = Wn + d * Em;
    #pragma unroll
    for (int e = 0; e < Em; ++e) { aR[e] += xv * wr[e]; aN[e] += xv * wn[e]; }
  }
  #pragma unroll
  for (int off = 16; off > 0; off >>= 1) {
    #pragma unroll
    for (int e = 0; e < Em; ++e) {
      aR[e] += __shfl_xor(aR[e], off, 32);
      aN[e] += __shfl_xor(aN[e], off, 32);
    }
  }
  if (lane == 0) {
    float ns[Em];
    #pragma unroll
    for (int e = 0; e < Em; ++e) {
      float nl = aN[e] + bn[e];
      float sp = (nl > 20.f) ? nl : log1pf(expf(nl));   // softplus
      ns[e] = aR[e] + br[e] + noise[(size_t)t * Em + e] * sp;
    }
    float v0 = -INFINITY, v1 = -INFINITY; int i0 = 0, i1 = 0;
    #pragma unroll
    for (int e = 0; e < Em; ++e) {
      float v = ns[e];
      if (v > v0) { v1 = v0; i1 = i0; v0 = v; i0 = e; }
      else if (v > v1) { v1 = v; i1 = e; }
    }
    float r  = expf(v1 - v0);                 // softmax over the two kept logits
    float g0 = 1.f / (1.f + r), g1 = r / (1.f + r);
    int elo, ehi; float glo, ghi;
    if (i0 < i1) { elo = i0; glo = g0; ehi = i1; ghi = g1; }
    else         { elo = i1; glo = g1; ehi = i0; ghi = g0; }
    int pid = elo * Em - (elo * (elo + 1)) / 2 + (ehi - elo - 1);   // 0..27
    int pos = atomicAdd(cnt + pid, 1);
    size_t o = (size_t)pid * CAPm + pos;
    tokL[o] = t; gloL[o] = glo; ghiL[o] = ghi;
  }
}

// Grouped GEMM over a 16-token tile of one expert-pair bucket. 8 waves / 256 threads.
__global__ void __launch_bounds__(256) moe_k(
    const float* __restrict__ x, const float* __restrict__ bd, const float* __restrict__ bu,
    const f16* __restrict__ WdP, const f16* __restrict__ WuP,
    const int* __restrict__ cnt, const int* __restrict__ tokL,
    const float* __restrict__ gloL, const float* __restrict__ ghiL,
    float* __restrict__ out) {
  __shared__ __align__(16) f16 xs[16][1032];   // 16 x 1024 f16, +8 pad (bank-conflict-free row reads)
  __shared__ __align__(16) f16 hs[16][520];    // 16 x 512 f16, +8 pad
  __shared__ int   tokS[16];
  __shared__ float gS[2][16];

  const int pid  = blockIdx.y;
  const int tile = blockIdx.x;
  const int n = cnt[pid];
  if (tile * 16 >= n) return;                  // uniform early-exit (worst-case grid)

  int elo = 0, rem = pid;
  while (rem >= 7 - elo) { rem -= 7 - elo; ++elo; }
  const int ehi = elo + 1 + rem;

  const int tid = threadIdx.x;
  if (tid < 16) {
    int idx = tile * 16 + tid;
    if (idx < n) {
      size_t o = (size_t)pid * CAPm + idx;
      tokS[tid] = tokL[o]; gS[0][tid] = gloL[o]; gS[1][tid] = ghiL[o];
    } else { tokS[tid] = -1; gS[0][tid] = 0.f; gS[1][tid] = 0.f; }
  }
  __syncthreads();

  { // stage 16 gathered token rows, f32 -> f16 LDS
    int r = tid >> 4, c = tid & 15;
    int tok = tokS[r];
    f16* dst = &xs[r][c * 64];
    if (tok >= 0) {
      const float4* src = (const float4*)(x + (size_t)tok * Dm + c * 64);
      #pragma unroll
      for (int i = 0; i < 16; ++i) {
        float4 v = src[i];
        dst[4*i] = (f16)v.x; dst[4*i+1] = (f16)v.y; dst[4*i+2] = (f16)v.z; dst[4*i+3] = (f16)v.w;
      }
    } else {
      #pragma unroll
      for (int i = 0; i < 64; ++i) dst[i] = (f16)0.f;
    }
  }
  __syncthreads();

  const int lane = tid & 31, wv = tid >> 5;
  const int hi = lane >> 4, lm = lane & 15;

  v8f acc[8];                                   // persistent C-fragments, 8 D-tiles per wave
  #pragma unroll
  for (int jj = 0; jj < 8; ++jj)
    #pragma unroll
    for (int j = 0; j < 8; ++j) acc[jj][j] = 0.f;

  for (int ei = 0; ei < 2; ++ei) {
    const int e = ei ? ehi : elo;
    if (ei) __syncthreads();                    // previous phase-B readers done before hs overwrite

    // Phase A: hs = gate * gelu(x @ Wd[e] + bd[e])   (32 N-tiles / 8 waves = 4 each)
    for (int jj = 0; jj < 4; ++jj) {
      const int ntile = jj * 8 + wv;
      const int ncol  = ntile * 16 + lm;
      v8f ha = {0.f,0.f,0.f,0.f,0.f,0.f,0.f,0.f};
      const f16* arow = &xs[lm][0];
      const f16* bp = WdP + (((size_t)e * KBD) * Hm + ncol) * 32;
      for (int kb = 0; kb < KBD; ++kb) {
        AFrag A, B;
        const int k0 = kb * 32 + hi * 8;
        A.h[0] = *(const v8h*)(arow + k0);      // k0..k0+7
        A.h[1] = *(const v8h*)(arow + k0 + 16); // k0+16..k0+23
        B.h[0] = *(const v8h*)(bp + hi * 8);
        B.h[1] = *(const v8h*)(bp + 16 + hi * 8);
        bp += (size_t)Hm * 32;
        ha = __builtin_amdgcn_wmma_f32_16x16x32_f16(false, A.v, false, B.v,
                                                    (short)0, ha, false, false);
      }
      const float bdv = bd[e * Hm + ncol];
      #pragma unroll
      for (int j = 0; j < 8; ++j) {
        const int m = j + 8 * hi;
        float v = ha[j] + bdv;
        v = 0.5f * v * (1.f + erff(v * 0.70710678118654752f));   // exact GELU
        hs[m][ncol] = (f16)(v * gS[ei][m]);
      }
    }
    __syncthreads();

    // Phase B: acc += hs @ Wu[e]   (64 D-tiles / 8 waves = 8 each)
    for (int jj = 0; jj < 8; ++jj) {
      const int dtile = jj * 8 + wv;
      const int dcol  = dtile * 16 + lm;
      const f16* arow = &hs[lm][0];
      const f16* bp = WuP + (((size_t)e * KBH) * Dm + dcol) * 32;
      v8f a2 = acc[jj];
      for (int kb = 0; kb < KBH; ++kb) {
        AFrag A, B;
        const int k0 = kb * 32 + hi * 8;
        A.h[0] = *(const v8h*)(arow + k0);
        A.h[1] = *(const v8h*)(arow + k0 + 16);
        B.h[0] = *(const v8h*)(bp + hi * 8);
        B.h[1] = *(const v8h*)(bp + 16 + hi * 8);
        bp += (size_t)Dm * 32;
        a2 = __builtin_amdgcn_wmma_f32_16x16x32_f16(false, A.v, false, B.v,
                                                    (short)0, a2, false, false);
      }
      acc[jj] = a2;
    }
  }

  // Writeout: add gated up-biases, plain stores (each token owned by exactly one block)
  #pragma unroll
  for (int jj = 0; jj < 8; ++jj) {
    const int dtile = jj * 8 + wv;
    const int dcol  = dtile * 16 + lm;
    const float bu0 = bu[elo * Dm + dcol];
    const float bu1 = bu[ehi * Dm + dcol];
    #pragma unroll
    for (int j = 0; j < 8; ++j) {
      const int m = j + 8 * hi;
      const int tok = tokS[m];
      if (tok >= 0)
        out[(size_t)tok * Dm + dcol] = acc[jj][j] + gS[0][m] * bu0 + gS[1][m] * bu1;
    }
  }
}

extern "C" void kernel_launch(void* const* d_in, const int* in_sizes, int n_in,
                              void* d_out, int out_size, void* d_ws, size_t ws_size,
                              hipStream_t stream) {
  (void)in_sizes; (void)n_in; (void)out_size; (void)ws_size;
  const float* x     = (const float*)d_in[0];
  const float* Wr    = (const float*)d_in[1];
  const float* br    = (const float*)d_in[2];
  const float* Wn    = (const float*)d_in[3];
  const float* bn    = (const float*)d_in[4];
  const float* Wd    = (const float*)d_in[5];
  const float* bd    = (const float*)d_in[6];
  const float* Wu    = (const float*)d_in[7];
  const float* bu    = (const float*)d_in[8];
  const float* noise = (const float*)d_in[9];
  float* out = (float*)d_out;

  char* ws = (char*)d_ws;
  int*   cnt  = (int*)  (ws + OFF_CNT);
  int*   tokL = (int*)  (ws + OFF_TOK);
  float* gloL = (float*)(ws + OFF_GLO);
  float* ghiL = (float*)(ws + OFF_GHI);
  f16*   WdP  = (f16*)  (ws + OFF_WDP);
  f16*   WuP  = (f16*)  (ws + OFF_WUP);

  hipLaunchKernelGGL(init_k, dim3(1), dim3(32), 0, stream, cnt);
  hipLaunchKernelGGL(pack_k, dim3((unsigned)((NWD + NWU) / 256)), dim3(256), 0, stream,
                     Wd, Wu, WdP, WuP);
  hipLaunchKernelGGL(router_k, dim3(NTOK / 8), dim3(256), 0, stream,
                     x, Wr, br, Wn, bn, noise, cnt, tokL, gloL, ghiL);
  hipLaunchKernelGGL(moe_k, dim3(MAXTILES, NPAIR), dim3(256), 0, stream,
                     x, bd, bu, WdP, WuP, cnt, tokL, gloL, ghiL, out);
}
